// Representation_39840116638271
// MI455X (gfx1250) — matrix operand order repaired and checked by reading, not compiled, not run on hardware
//
#include <hip/hip_runtime.h>

#define DEV __device__ __forceinline__

typedef _Float16 v8h  __attribute__((ext_vector_type(8)));
typedef _Float16 v16h __attribute__((ext_vector_type(16)));
typedef float    v8f  __attribute__((ext_vector_type(8)));

constexpr int kN = 50000, kE = 800000, kB = 64, kD = 64, kH = 2,
              kG = 50, kHID = 128, kL = 3;
constexpr int kNp = ((kN + 127) / 128) * 128;  // 50048, padded node rows
constexpr float kCUT = 8.0f;
constexpr float kPI  = 3.14159265358979323846f;
static_assert(kE % 128 == 0, "edge tiles assumed full");
static_assert(kNp % 128 == 0, "node tiles assumed full");

DEV float sspf(float x) {
  // softplus(x) - log(2), numerically stable
  return fmaxf(x, 0.0f) + log1pf(__expf(-fabsf(x))) - 0.69314718055994531f;
}
DEV int   f2key(float f) { int i = __float_as_int(f); return i >= 0 ? i : ~i; }
DEV float key2f(int k)   { return __int_as_float(k >= 0 ? k : ~k); }

// ---- CDNA5 async global->LDS copy (ASYNCcnt-tracked), ISA 08 §4 ----
DEV void async_b128(unsigned lds_byte_addr, const void* gptr) {
  asm volatile("global_load_async_to_lds_b128 %0, %1, off"
               :: "v"(lds_byte_addr), "v"(gptr)
               : "memory");
}
DEV void async_wait0() {
  asm volatile("s_wait_asynccnt 0" ::: "memory");
}

DEV v16h frag2(v8h lo, v8h hi) {
  union { v16h v; v8h h[2]; } u; u.h[0] = lo; u.h[1] = hi; return u.v;
}
// A fragment (16xK tile, rows row0..+15, K chunk kbase..+31), ISA 7.12.2:
// lane m = lane&15; elem i<8 -> K = kbase+8*sel+i ; i>=8 -> K = kbase+16+8*sel+(i-8)
DEV v16h loadA(const _Float16* base, int row0, int kbase, int ld, int fn, int sel) {
  const _Float16* p = base + (row0 + fn) * ld + kbase + 8 * sel;
  return frag2(*(const v8h*)p, *(const v8h*)(p + 16));
}
// B fragment from TRANSPOSED weights Wt[out][in_pad] (row-major, ld = in_pad):
// lane n = lane&15; elem i<8 -> K = kbase+8*sel+i ; i>=8 -> K = kbase+16+8*sel+(i-8)
DEV v16h loadBT(const _Float16* wt, int col0, int kbase, int ld, int fn, int sel) {
  const _Float16* p = wt + (col0 + fn) * ld + kbase + 8 * sel;
  return frag2(*(const v8h*)p, *(const v8h*)(p + 16));
}
DEV v8f wmma16(v16h a, v16h b, v8f c) {
  return __builtin_amdgcn_wmma_f32_16x16x32_f16(false, a, false, b, (short)0, c,
                                                false, false);
}

// ---------------- utility kernels ----------------
__global__ void k_fill_f32(float* p, float v, int n) {
  int t = blockIdx.x * 256 + threadIdx.x;
  if (t < n) p[t] = v;
}
__global__ void k_fill_i32(int* p, int v, int n) {
  int t = blockIdx.x * 256 + threadIdx.x;
  if (t < n) p[t] = v;
}
// convert f32 weight (in_dim x out_dim) -> f16 transposed+padded (out_dim x in_pad)
__global__ void k_wt(const float* src, _Float16* dst, int in_dim, int out_dim,
                     int in_pad) {
  int t = blockIdx.x * 256 + threadIdx.x;
  if (t >= out_dim * in_pad) return;
  int o = t / in_pad, i = t % in_pad;
  float v = (i < in_dim) ? src[i * out_dim + o] : 0.0f;
  dst[o * in_pad + i] = (_Float16)v;
}
// x (f16, padded rows) = emb_table[Z]
__global__ void k_embed(const int* Z, const float* emb, _Float16* x) {
  int t = blockIdx.x * 256 + threadIdx.x;
  if (t >= kNp * kD) return;
  int n = t >> 6, d = t & 63;
  float v = (n < kN) ? emb[Z[n] * kD + d] : 0.0f;
  x[t] = (_Float16)v;
}
__global__ void k_geom(const float* R, const int* src, const int* dst, float* rb,
                       float* cb) {
  int e = blockIdx.x * 256 + threadIdx.x;
  if (e >= kE) return;
  int s = src[e], d = dst[e];
  float dx = R[s * 3 + 0] - R[d * 3 + 0];
  float dy = R[s * 3 + 1] - R[d * 3 + 1];
  float dz = R[s * 3 + 2] - R[d * 3 + 2];
  float r = sqrtf(dx * dx + dy * dy + dz * dz);
  float c = 0.5f * (cosf(r * (kPI / kCUT)) + 1.0f) * (r < kCUT ? 1.0f : 0.0f);
  rb[e] = r;
  cb[e] = c;
}

// ------- node GEMM: h = x @ fc_w (Np x 64 f16 -> Np x 128 f32), async-staged ----
__global__ void k_nodegemm_h(const _Float16* x, const _Float16* fcwt, float* hb) {
  __shared__ __align__(16) _Float16 xt[128 * 64];
  int tid = threadIdx.x;
  size_t row0 = (size_t)blockIdx.x * 128;
  {  // 16KB contiguous tile: thread t moves 64B via 4 async b128 ops
    const char* g = (const char*)(x + row0 * 64) + tid * 64;
    unsigned l = (unsigned)(size_t)((char*)xt + tid * 64);
    for (int i = 0; i < 4; ++i) async_b128(l + i * 16, g + i * 16);
    async_wait0();
  }
  __syncthreads();
  int wv = tid >> 5, lane = tid & 31, fn = lane & 15, sel = lane >> 4;
  int wr = wv * 16;
  for (int ct = 0; ct < 8; ++ct) {
    v8f acc = {};
    for (int kb = 0; kb < 64; kb += 32)
      acc = wmma16(loadA(xt, wr, kb, 64, fn, sel),
                   loadBT(fcwt, ct * 16, kb, 64, fn, sel), acc);
    int col = ct * 16 + fn;
    for (int r = 0; r < 8; ++r)
      hb[(row0 + wr + r + 8 * sel) * 128 + col] = acc[r];
  }
}

// ---------------- attention scalars ----------------
__global__ void k_attn(const float* hb, const float* al, const float* ar,
                       float* el, float* er) {
  int t = blockIdx.x * 256 + threadIdx.x;
  if (t >= kN * kH) return;
  int n = t >> 1, hh = t & 1;
  const float* hr = hb + (size_t)n * 128 + hh * 64;
  float sl = 0.f, sr = 0.f;
  for (int d = 0; d < 64; ++d) {
    float v = hr[d];
    sl += v * al[hh * 64 + d];
    sr += v * ar[hh * 64 + d];
  }
  el[t] = sl;
  er[t] = sr;
}
__global__ void k_edge_e(const int* src, const int* dst, const float* el,
                         const float* er, float* ebuf, int* mkey) {
  int t = blockIdx.x * 256 + threadIdx.x;
  if (t >= kE * kH) return;
  int e = t >> 1, hh = t & 1;
  float v = el[src[e] * 2 + hh] + er[dst[e] * 2 + hh];
  v = v < 0.0f ? 0.2f * v : v;
  ebuf[t] = v;
  atomicMax(&mkey[dst[e] * 2 + hh], f2key(v));
}
__global__ void k_edge_ex(const int* dst, const int* mkey, float* ebuf,
                          float* denom) {
  int t = blockIdx.x * 256 + threadIdx.x;
  if (t >= kE * kH) return;
  int e = t >> 1, hh = t & 1;
  float m = key2f(mkey[dst[e] * 2 + hh]);
  float ex = __expf(ebuf[t] - m);
  ebuf[t] = ex;
  atomicAdd(&denom[dst[e] * 2 + hh], ex);
}
__global__ void k_edge_s(const int* dst, const float* denom, const float* cb,
                         float* ebuf) {
  int t = blockIdx.x * 256 + threadIdx.x;
  if (t >= kE * kH) return;
  int e = t >> 1, hh = t & 1;
  ebuf[t] = ebuf[t] / denom[dst[e] * 2 + hh] * cb[e];
}

// ------- fused edge filter: gs -> W1 -> ssp -> W2, * h[src] * (alpha*C),
//         atomic scatter into agg (Np x 128). 128 edges/block, 8 waves. -------
__global__ void k_edge_filter(const int* src, const int* dst, const float* rb,
                              const float* offs, const float* wids,
                              const float* sbuf, const float* hb,
                              const _Float16* w1t, const float* b1,
                              const _Float16* w2t, const float* b2, float* agg) {
  __shared__ __align__(16) _Float16 gsA[128 * 72];
  __shared__ __align__(16) _Float16 hid[128 * 136];
  __shared__ float s_sc[128 * 2];
  __shared__ int s_src[128], s_dst[128];
  __shared__ float s_cf[64], s_of[64];
  int tid = threadIdx.x;
  int e0 = blockIdx.x * 128;  // kE % 128 == 0 -> every tile full
  if (tid < 64) {
    if (tid < kG) {
      float wd = wids[tid];
      s_cf[tid] = -0.5f / (wd * wd);
      s_of[tid] = offs[tid];
    } else {
      s_cf[tid] = 0.0f;
      s_of[tid] = 0.0f;
    }
  }
  if (tid < 128) {
    int e = e0 + tid;
    int sn = src[e];
    s_src[tid] = sn;
    s_dst[tid] = dst[e];
    s_sc[tid * 2 + 0] = sbuf[e * 2 + 0];
    s_sc[tid * 2 + 1] = sbuf[e * 2 + 1];
    __builtin_prefetch(hb + (size_t)sn * 128, 0, 0);  // global_prefetch
  }
  __syncthreads();
  {  // Gaussian smearing into A-tile (f16), padded 50 -> 64
    int eL = tid >> 1, g0 = (tid & 1) * 32;
    float r = rb[e0 + eL];
    for (int g = g0; g < g0 + 32; ++g) {
      float gv = 0.0f;
      if (g < kG) {
        float d = r - s_of[g];
        gv = __expf(s_cf[g] * d * d);
      }
      gsA[eL * 72 + g] = (_Float16)gv;
    }
  }
  __syncthreads();
  int wv = tid >> 5, lane = tid & 31, fn = lane & 15, sel = lane >> 4;
  int wr = wv * 16;
  // GEMM1: hid = ssp(gs @ W1 + b1)   (128 x 64) @ (64 x 128)
  for (int ct = 0; ct < 8; ++ct) {
    v8f acc = {};
    for (int kb = 0; kb < 64; kb += 32)
      acc = wmma16(loadA(gsA, wr, kb, 72, fn, sel),
                   loadBT(w1t, ct * 16, kb, 64, fn, sel), acc);
    int col = ct * 16 + fn;
    float bias = b1[col];
    for (int r = 0; r < 8; ++r)
      hid[(wr + r + 8 * sel) * 136 + col] = (_Float16)sspf(acc[r] + bias);
  }
  __syncthreads();
  // GEMM2 + message + scatter: (128 x 128) @ (128 x 128)
  for (int ct = 0; ct < 8; ++ct) {
    v8f acc = {};
    for (int kb = 0; kb < 128; kb += 32)
      acc = wmma16(loadA(hid, wr, kb, 136, fn, sel),
                   loadBT(w2t, ct * 16, kb, 128, fn, sel), acc);
    int col = ct * 16 + fn;
    float bias = b2[col];
    int head = col >> 6;
    for (int r = 0; r < 8; ++r) {
      int eL = wr + r + 8 * sel;
      float wval = acc[r] + bias;
      float hval = hb[(size_t)s_src[eL] * 128 + col];
      float contrib = hval * wval * s_sc[eL * 2 + head];
      atomicAdd(&agg[(size_t)s_dst[eL] * 128 + col], contrib);
    }
  }
}

// ------- node update: x += mlp2(ssp(mlp1(ssp(agg)))), x is f16 -------
__global__ void k_update(const float* agg, const _Float16* m1t, const float* b1,
                         const _Float16* m2t, const float* b2, _Float16* x) {
  __shared__ __align__(16) _Float16 t1[128 * 136];
  __shared__ __align__(16) _Float16 t2[128 * 72];
  int tid = threadIdx.x;
  size_t row0 = (size_t)blockIdx.x * 128;
  for (int i = tid; i < 128 * 128; i += 256) {
    int r = i >> 7, c = i & 127;
    t1[r * 136 + c] = (_Float16)sspf(agg[(row0 + r) * 128 + c]);
  }
  __syncthreads();
  int wv = tid >> 5, lane = tid & 31, fn = lane & 15, sel = lane >> 4;
  int wr = wv * 16;
  for (int ct = 0; ct < 4; ++ct) {
    v8f acc = {};
    for (int kb = 0; kb < 128; kb += 32)
      acc = wmma16(loadA(t1, wr, kb, 136, fn, sel),
                   loadBT(m1t, ct * 16, kb, 128, fn, sel), acc);
    int col = ct * 16 + fn;
    float bias = b1[col];
    for (int r = 0; r < 8; ++r)
      t2[(wr + r + 8 * sel) * 72 + col] = (_Float16)sspf(acc[r] + bias);
  }
  __syncthreads();
  for (int ct = 0; ct < 4; ++ct) {
    v8f acc = {};
    for (int kb = 0; kb < 64; kb += 32)
      acc = wmma16(loadA(t2, wr, kb, 72, fn, sel),
                   loadBT(m2t, ct * 16, kb, 64, fn, sel), acc);
    int col = ct * 16 + fn;
    float bias = b2[col];
    for (int r = 0; r < 8; ++r) {
      size_t idx = (row0 + wr + r + 8 * sel) * 64 + col;
      x[idx] = (_Float16)((float)x[idx] + acc[r] + bias);
    }
  }
}

// ------- output MLP + segment-sum over graphs -------
__global__ void k_out(const _Float16* x, const _Float16* o1t, const float* b1,
                      const float* w2, const float* b2, const int* gid,
                      float* out) {
  __shared__ __align__(16) _Float16 xt[128 * 64];
  int tid = threadIdx.x;
  size_t row0 = (size_t)blockIdx.x * 128;
  {
    const char* g = (const char*)(x + row0 * 64) + tid * 64;
    unsigned l = (unsigned)(size_t)((char*)xt + tid * 64);
    for (int i = 0; i < 4; ++i) async_b128(l + i * 16, g + i * 16);
    async_wait0();
  }
  __syncthreads();
  int wv = tid >> 5, lane = tid & 31, fn = lane & 15, sel = lane >> 4;
  int wr = wv * 16;
  float part[8] = {0, 0, 0, 0, 0, 0, 0, 0};
  for (int ct = 0; ct < 8; ++ct) {
    v8f acc = {};
    for (int kb = 0; kb < 64; kb += 32)
      acc = wmma16(loadA(xt, wr, kb, 64, fn, sel),
                   loadBT(o1t, ct * 16, kb, 64, fn, sel), acc);
    int col = ct * 16 + fn;
    float bias = b1[col], w2c = w2[col];
    for (int r = 0; r < 8; ++r) part[r] += sspf(acc[r] + bias) * w2c;
  }
  for (int r = 0; r < 8; ++r) {
    float s = part[r];
    for (int off = 8; off > 0; off >>= 1) s += __shfl_xor(s, off, 16);
    if (fn == 0) {
      size_t row = row0 + wr + r + 8 * sel;
      if (row < (size_t)kN) atomicAdd(&out[gid[row]], s + b2[0]);
    }
  }
}

extern "C" void kernel_launch(void* const* d_in, const int* in_sizes, int n_in,
                              void* d_out, int out_size, void* d_ws,
                              size_t ws_size, hipStream_t stream) {
  const float* R      = (const float*)d_in[0];
  const int*   Z      = (const int*)d_in[1];
  const int*   src    = (const int*)d_in[2];
  const int*   dst    = (const int*)d_in[3];
  const int*   gid    = (const int*)d_in[4];
  const float* emb    = (const float*)d_in[5];
  const float* offs   = (const float*)d_in[6];
  const float* wids   = (const float*)d_in[7];
  const float* fc_w   = (const float*)d_in[8];
  const float* attn_l = (const float*)d_in[9];
  const float* attn_r = (const float*)d_in[10];
  const float* fw1    = (const float*)d_in[11];
  const float* fb1    = (const float*)d_in[12];
  const float* fw2    = (const float*)d_in[13];
  const float* fb2    = (const float*)d_in[14];
  const float* mw1    = (const float*)d_in[15];
  const float* mb1    = (const float*)d_in[16];
  const float* mw2    = (const float*)d_in[17];
  const float* mb2    = (const float*)d_in[18];
  const float* ow1    = (const float*)d_in[19];
  const float* ob1    = (const float*)d_in[20];
  const float* ow2    = (const float*)d_in[21];
  const float* ob2    = (const float*)d_in[22];
  float* out = (float*)d_out;

  size_t off = 0;
  auto take = [&](size_t bytes) -> char* {
    char* p = (char*)d_ws + off;
    off += (bytes + 255) & ~(size_t)255;
    return p;
  };
  _Float16* xbuf  = (_Float16*)take((size_t)kNp * 64 * 2);
  float*    hbuf  = (float*)take((size_t)kNp * 128 * 4);
  float*    aggb  = (float*)take((size_t)kNp * 128 * 4);
  float*    elb   = (float*)take((size_t)kN * 2 * 4);
  float*    erb   = (float*)take((size_t)kN * 2 * 4);
  int*      mkey  = (int*)take((size_t)kN * 2 * 4);
  float*    denom = (float*)take((size_t)kN * 2 * 4);
  float*    ebuf  = (float*)take((size_t)kE * 2 * 4);
  float*    rb    = (float*)take((size_t)kE * 4);
  float*    cbuf  = (float*)take((size_t)kE * 4);
  _Float16* fcwt  = (_Float16*)take((size_t)kL * 128 * 64 * 2);
  _Float16* fw1t  = (_Float16*)take((size_t)kL * 128 * 64 * 2);
  _Float16* fw2t  = (_Float16*)take((size_t)kL * 128 * 128 * 2);
  _Float16* mw1t  = (_Float16*)take((size_t)kL * 64 * 128 * 2);
  _Float16* mw2t  = (_Float16*)take((size_t)kL * 64 * 64 * 2);
  _Float16* ow1t  = (_Float16*)take((size_t)128 * 64 * 2);

  dim3 b256(256);
  auto g1 = [](int n) { return dim3((unsigned)((n + 255) / 256)); };

  for (int l = 0; l < kL; ++l) {
    k_wt<<<g1(128 * 64), b256, 0, stream>>>(fc_w + l * 64 * 128,
                                            fcwt + l * 128 * 64, 64, 128, 64);
    k_wt<<<g1(128 * 64), b256, 0, stream>>>(fw1 + l * kG * 128,
                                            fw1t + l * 128 * 64, kG, 128, 64);
    k_wt<<<g1(128 * 128), b256, 0, stream>>>(fw2 + l * 128 * 128,
                                             fw2t + l * 128 * 128, 128, 128, 128);
    k_wt<<<g1(64 * 128), b256, 0, stream>>>(mw1 + l * 128 * 64,
                                            mw1t + l * 64 * 128, 128, 64, 128);
    k_wt<<<g1(64 * 64), b256, 0, stream>>>(mw2 + l * 64 * 64,
                                           mw2t + l * 64 * 64, 64, 64, 64);
  }
  k_wt<<<g1(128 * 64), b256, 0, stream>>>(ow1, ow1t, 64, 128, 64);
  k_embed<<<g1(kNp * 64), b256, 0, stream>>>(Z, emb, xbuf);
  k_geom<<<g1(kE), b256, 0, stream>>>(R, src, dst, rb, cbuf);
  k_fill_f32<<<1, b256, 0, stream>>>(out, 0.0f, kB);

  const int NBN = kNp / 128, NBE = kE / 128;
  for (int l = 0; l < kL; ++l) {
    k_nodegemm_h<<<NBN, b256, 0, stream>>>(xbuf, fcwt + l * 128 * 64, hbuf);
    k_attn<<<g1(kN * 2), b256, 0, stream>>>(hbuf, attn_l + l * kH * kD,
                                            attn_r + l * kH * kD, elb, erb);
    k_fill_i32<<<g1(kN * 2), b256, 0, stream>>>(mkey, (int)0x80000000, kN * 2);
    k_fill_f32<<<g1(kN * 2), b256, 0, stream>>>(denom, 0.0f, kN * 2);
    k_fill_f32<<<g1(kNp * 128), b256, 0, stream>>>(aggb, 0.0f, kNp * 128);
    k_edge_e<<<g1(kE * 2), b256, 0, stream>>>(src, dst, elb, erb, ebuf, mkey);
    k_edge_ex<<<g1(kE * 2), b256, 0, stream>>>(dst, mkey, ebuf, denom);
    k_edge_s<<<g1(kE * 2), b256, 0, stream>>>(dst, denom, cbuf, ebuf);
    k_edge_filter<<<NBE, b256, 0, stream>>>(src, dst, rb, offs, wids, ebuf, hbuf,
                                            fw1t + l * 128 * 64, fb1 + l * kHID,
                                            fw2t + l * 128 * 128, fb2 + l * 128,
                                            aggb);
    k_update<<<NBN, b256, 0, stream>>>(aggb, mw1t + l * 64 * 128, mb1 + l * 64,
                                       mw2t + l * 64 * 64, mb2 + l * 64, xbuf);
  }
  k_out<<<NBN, b256, 0, stream>>>(xbuf, ow1t, ob1, ow2, ob2, gid, out);
}
